// DGHANlayer_29978871726245
// MI455X (gfx1250) — matrix operand order repaired and checked by reading, not compiled, not run on hardware
//
#include <hip/hip_runtime.h>
#include <hip/hip_bf16.h>

// ---------------------------------------------------------------------------
// DGHAN layer: two GATConv(H=2, C=64) over different edge sets, ELU, average.
// N=50000 nodes, E=800000 edges, IN_C=128, H*OUT_C=128.
//
// GEMM (x@W, 3.3 GFLOP total) via v_wmma_f32_16x16x32_bf16: W is pre-converted
// once to transposed bf16 (L2-resident, 64KB), x tiles staged as bf16 in LDS,
// so the WMMA loop is pure ds_load_b128 / global_load_b128 / v_wmma.
// Edge softmax+aggregation is the real cost: ~1.6 GB gather+atomic traffic
// -> ~70us at 23.3 TB/s HBM; xw (25.6MB/conv) stays in the 192MB L2.
// ---------------------------------------------------------------------------

#define N_NODES 50000
#define E_EDGES 800000
#define IN_C    128
#define OUT_HC  128      // HEADS * OUT_C
#define HEADS   2
#define OUT_C   64
#define NEG_SLOPE 0.2f

typedef __attribute__((ext_vector_type(16))) __bf16 v16bf;
typedef __attribute__((ext_vector_type(8)))  __bf16 v8bf;
typedef __attribute__((ext_vector_type(4)))  __bf16 v4bf;
typedef __attribute__((ext_vector_type(8)))  float  v8f;

// ---- helpers --------------------------------------------------------------

__device__ __forceinline__ __bf16 f2bf(float f) {
    unsigned u = __float_as_uint(f);
    unsigned r = u + 0x7fffu + ((u >> 16) & 1u);     // round-to-nearest-even
    return __builtin_bit_cast(__bf16, (unsigned short)(r >> 16));
}

// order-preserving float <-> uint encoding (for atomicMax on floats)
__device__ __forceinline__ unsigned f2ord(float f) {
    unsigned u = __float_as_uint(f);
    return (u & 0x80000000u) ? ~u : (u | 0x80000000u);
}
__device__ __forceinline__ float ord2f(unsigned u) {
    return (u & 0x80000000u) ? __uint_as_float(u & 0x7fffffffu)
                             : __uint_as_float(~u);
}
#define NEG_INF_ORD 0x007FFFFFu   // f2ord(-inf)

__device__ __forceinline__ float lrelu(float v) {
    return v > 0.0f ? v : NEG_SLOPE * v;
}

// ---- K0a: zero / init workspace -------------------------------------------

__global__ void init_kernel(float* acc_pc, float* acc_mc,
                            float* denom_pc, float* denom_mc,
                            unsigned* m_pc, unsigned* m_mc) {
    const long long total = (long long)N_NODES * OUT_HC;
    for (long long i = blockIdx.x * (long long)blockDim.x + threadIdx.x;
         i < total; i += (long long)gridDim.x * blockDim.x) {
        acc_pc[i] = 0.0f;
        acc_mc[i] = 0.0f;
        if (i < (long long)N_NODES * HEADS) {
            denom_pc[i] = 0.0f;
            denom_mc[i] = 0.0f;
            m_pc[i] = NEG_INF_ORD;
            m_mc[i] = NEG_INF_ORD;
        }
    }
}

// ---- K0b: pre-convert W to transposed bf16: Wt[conv][col][K] --------------
// grid (128 cols, 2 convs), block 128 (K). 64KB output, L2-resident forever.

__global__ void wt_kernel(const float* __restrict__ W_pc,
                          const float* __restrict__ W_mc,
                          __bf16* __restrict__ wt) {
    const int col  = blockIdx.x;
    const int conv = blockIdx.y;
    const int K    = threadIdx.x;
    const float* W = conv ? W_mc : W_pc;
    wt[(size_t)conv * (IN_C * OUT_HC) + (size_t)col * IN_C + K] =
        f2bf(W[K * OUT_HC + col]);
}

// ---- K1: xw = x @ W via WMMA bf16 -----------------------------------------
// block = 256 threads (8 waves). Each block: 16 rows of x, all 128 out cols.
// Wave w owns columns [16w, 16w+16). gridDim.x = N/16, gridDim.y = conv.
// x tile staged as bf16 in LDS (stride 136 -> 272B rows, 16B aligned).

__global__ void __launch_bounds__(256)
gemm_xw_kernel(const float* __restrict__ x,
               const __bf16* __restrict__ wt,
               float* __restrict__ xw_pc,
               float* __restrict__ xw_mc) {
    __shared__ __bf16 lxb[16][136];

    float* __restrict__ xw = blockIdx.y ? xw_mc : xw_pc;

    const int rowBase = blockIdx.x * 16;
    const int tid  = threadIdx.x;

    // stage 16x128 tile of x into LDS as bf16 (one conversion per element)
    for (int i = tid; i < 16 * 32; i += 256) {
        const int r = i >> 5, cq = i & 31;
        const float4 xv = *(const float4*)(x + (long long)(rowBase + r) * IN_C + cq * 4);
        v4bf p = { f2bf(xv.x), f2bf(xv.y), f2bf(xv.z), f2bf(xv.w) };
        *(v4bf*)&lxb[r][cq * 4] = p;
    }
    __syncthreads();

    const int wave = tid >> 5;                  // 0..7 -> column tile
    const int lane = tid & 31;
    const int half = lane >> 4;                 // 0 or 1
    const int m    = lane & 15;                 // row (A) / col-in-tile (B,D)
    const int col  = wave * 16 + m;             // global output column

    const __bf16* __restrict__ wtp =
        wt + (size_t)blockIdx.y * (IN_C * OUT_HC) + (size_t)col * IN_C;

    v8f c = {};
    #pragma unroll
    for (int kk = 0; kk < 4; ++kk) {            // K = 128 in steps of 32
        const int kb = kk * 32;
        // A 16x32 bf16 layout: lanes 0-15 hold K = kb+{0..7, 16..23},
        // lanes 16-31 hold K = kb+{8..15, 24..31} -> two 16B runs per lane
        v8bf a0 = *(const v8bf*)&lxb[m][kb + half * 8];
        v8bf a1 = *(const v8bf*)&lxb[m][kb + half * 8 + 16];
        v16bf a = __builtin_shufflevector(a0, a1,
                      0,1,2,3,4,5,6,7,8,9,10,11,12,13,14,15);
        // B 32x16 bf16 layout: lane holds K = kb + 8*half + {0..7, 16..23}
        // = two 8-bf16 runs in transposed Wt -> two global b128 loads
        v8bf b0 = *(const v8bf*)(wtp + kb + half * 8);
        v8bf b1 = *(const v8bf*)(wtp + kb + half * 8 + 16);
        v16bf b = __builtin_shufflevector(b0, b1,
                      0,1,2,3,4,5,6,7,8,9,10,11,12,13,14,15);
        c = __builtin_amdgcn_wmma_f32_16x16x32_bf16(
                /*neg_a=*/false, a, /*neg_b=*/false, b,
                /*c_mod=*/(short)0, c, /*reuse_a=*/false, /*reuse_b=*/false);
    }

    // D layout: VGPR r -> row (r + 8*half), col = lane&15 within tile
    #pragma unroll
    for (int r = 0; r < 8; ++r) {
        const int row = rowBase + half * 8 + r;
        xw[(long long)row * OUT_HC + col] = c[r];
    }
}

// ---- K2: per-node attention logits a_src, a_dst ---------------------------
// wave per node; lane l handles channels 4l..4l+3 (head = l/16).
// att_{src,dst} flat [H*C]=[128] matches the channel layout exactly.

__global__ void __launch_bounds__(256)
att_kernel(const float* __restrict__ xw_pc, const float* __restrict__ xw_mc,
           const float* __restrict__ attS_pc, const float* __restrict__ attD_pc,
           const float* __restrict__ attS_mc, const float* __restrict__ attD_mc,
           float* __restrict__ as_pc, float* __restrict__ ad_pc,
           float* __restrict__ as_mc, float* __restrict__ ad_mc) {
    const int conv = blockIdx.y;
    const float* __restrict__ xw   = conv ? xw_mc   : xw_pc;
    const float* __restrict__ attS = conv ? attS_mc : attS_pc;
    const float* __restrict__ attD = conv ? attD_mc : attD_pc;
    float* __restrict__ as = conv ? as_mc : as_pc;
    float* __restrict__ ad = conv ? ad_mc : ad_pc;

    const int node = blockIdx.x * 8 + (threadIdx.x >> 5);
    if (node >= N_NODES) return;
    const int lane = threadIdx.x & 31;

    const float4 xv = *(const float4*)(xw + (long long)node * OUT_HC + lane * 4);
    const float4 sv = *(const float4*)(attS + lane * 4);
    const float4 dv = *(const float4*)(attD + lane * 4);

    float ps = xv.x * sv.x + xv.y * sv.y + xv.z * sv.z + xv.w * sv.w;
    float pd = xv.x * dv.x + xv.y * dv.y + xv.z * dv.z + xv.w * dv.w;
    // reduce within each 16-lane half (one head per half)
    #pragma unroll
    for (int msk = 8; msk >= 1; msk >>= 1) {
        ps += __shfl_xor(ps, msk, 32);
        pd += __shfl_xor(pd, msk, 32);
    }
    if ((lane & 15) == 0) {
        const int h = lane >> 4;
        as[(long long)node * HEADS + h] = ps;
        ad[(long long)node * HEADS + h] = pd;
    }
}

// ---- K3: segment max of edge logits (ordered-uint atomicMax) --------------

__global__ void edge_max_kernel(const long long* __restrict__ ei_pc,
                                const long long* __restrict__ ei_mc,
                                const float* __restrict__ as_pc,
                                const float* __restrict__ ad_pc,
                                const float* __restrict__ as_mc,
                                const float* __restrict__ ad_mc,
                                unsigned* __restrict__ m_pc,
                                unsigned* __restrict__ m_mc) {
    const int conv = blockIdx.y;
    const long long* ei = conv ? ei_mc : ei_pc;
    const float* as = conv ? as_mc : as_pc;
    const float* ad = conv ? ad_mc : ad_pc;
    unsigned* m = conv ? m_mc : m_pc;

    const int e = blockIdx.x * blockDim.x + threadIdx.x;
    if (e >= E_EDGES) return;
    const long long s = ei[e], d = ei[E_EDGES + e];
    #pragma unroll
    for (int h = 0; h < HEADS; ++h) {
        const float v = lrelu(as[s * HEADS + h] + ad[d * HEADS + h]);
        atomicMax(&m[d * HEADS + h], f2ord(v));
    }
}

// ---- K4: segment sum of exp(e - m[dst]) -----------------------------------

__global__ void edge_sum_kernel(const long long* __restrict__ ei_pc,
                                const long long* __restrict__ ei_mc,
                                const float* __restrict__ as_pc,
                                const float* __restrict__ ad_pc,
                                const float* __restrict__ as_mc,
                                const float* __restrict__ ad_mc,
                                const unsigned* __restrict__ m_pc,
                                const unsigned* __restrict__ m_mc,
                                float* __restrict__ den_pc,
                                float* __restrict__ den_mc) {
    const int conv = blockIdx.y;
    const long long* ei = conv ? ei_mc : ei_pc;
    const float* as = conv ? as_mc : as_pc;
    const float* ad = conv ? ad_mc : ad_pc;
    const unsigned* m = conv ? m_mc : m_pc;
    float* den = conv ? den_mc : den_pc;

    const int e = blockIdx.x * blockDim.x + threadIdx.x;
    if (e >= E_EDGES) return;
    const long long s = ei[e], d = ei[E_EDGES + e];
    #pragma unroll
    for (int h = 0; h < HEADS; ++h) {
        const float v  = lrelu(as[s * HEADS + h] + ad[d * HEADS + h]);
        const float mv = ord2f(m[d * HEADS + h]);   // finite: d has >=1 edge
        atomicAdd(&den[d * HEADS + h], __expf(v - mv));
    }
}

// ---- K5: message scatter: acc[dst] += alpha * xw[src] ---------------------
// one wave per edge; lane l handles channels 4l..4l+3 (head = l/16).

__global__ void __launch_bounds__(256)
edge_msg_kernel(const long long* __restrict__ ei_pc,
                const long long* __restrict__ ei_mc,
                const float* __restrict__ as_pc,
                const float* __restrict__ ad_pc,
                const float* __restrict__ as_mc,
                const float* __restrict__ ad_mc,
                const unsigned* __restrict__ m_pc,
                const unsigned* __restrict__ m_mc,
                const float* __restrict__ den_pc,
                const float* __restrict__ den_mc,
                const float* __restrict__ xw_pc,
                const float* __restrict__ xw_mc,
                float* __restrict__ acc_pc,
                float* __restrict__ acc_mc) {
    const int conv = blockIdx.y;
    const long long* ei = conv ? ei_mc : ei_pc;
    const float* as = conv ? as_mc : as_pc;
    const float* ad = conv ? ad_mc : ad_pc;
    const unsigned* m = conv ? m_mc : m_pc;
    const float* den = conv ? den_mc : den_pc;
    const float* xw = conv ? xw_mc : xw_pc;
    float* acc = conv ? acc_mc : acc_pc;

    const int e = blockIdx.x * 8 + (threadIdx.x >> 5);
    if (e >= E_EDGES) return;
    const int lane = threadIdx.x & 31;
    const int h = lane >> 4;

    const long long s = ei[e], d = ei[E_EDGES + e];
    const float v  = lrelu(as[s * HEADS + h] + ad[d * HEADS + h]);
    const float mv = ord2f(m[d * HEADS + h]);
    const float alpha = __expf(v - mv) / (den[d * HEADS + h] + 1e-16f);

    const float4 xv = *(const float4*)(xw + s * OUT_HC + lane * 4);
    float* ap = acc + d * OUT_HC + lane * 4;
    atomicAdd(ap + 0, xv.x * alpha);
    atomicAdd(ap + 1, xv.y * alpha);
    atomicAdd(ap + 2, xv.z * alpha);
    atomicAdd(ap + 3, xv.w * alpha);
}

// ---- K6: epilogue: 0.5*(elu(acc_pc+b_pc) + elu(acc_mc+b_mc)) --------------

__global__ void final_kernel(const float* __restrict__ acc_pc,
                             const float* __restrict__ acc_mc,
                             const float* __restrict__ bias_pc,
                             const float* __restrict__ bias_mc,
                             float* __restrict__ out) {
    const long long total = (long long)N_NODES * OUT_HC;
    const long long i = blockIdx.x * (long long)blockDim.x + threadIdx.x;
    if (i >= total) return;
    const int c = (int)(i & (OUT_HC - 1));
    float a = acc_pc[i] + bias_pc[c];
    float b = acc_mc[i] + bias_mc[c];
    a = a > 0.0f ? a : expm1f(a);
    b = b > 0.0f ? b : expm1f(b);
    out[i] = 0.5f * (a + b);
}

// ---------------------------------------------------------------------------

extern "C" void kernel_launch(void* const* d_in, const int* in_sizes, int n_in,
                              void* d_out, int out_size, void* d_ws, size_t ws_size,
                              hipStream_t stream) {
    const float*     x        = (const float*)d_in[0];
    const long long* ei_pc    = (const long long*)d_in[1];
    const long long* ei_mc    = (const long long*)d_in[2];
    const float*     W_pc     = (const float*)d_in[3];
    const float*     attS_pc  = (const float*)d_in[4];
    const float*     attD_pc  = (const float*)d_in[5];
    const float*     bias_pc  = (const float*)d_in[6];
    const float*     W_mc     = (const float*)d_in[7];
    const float*     attS_mc  = (const float*)d_in[8];
    const float*     attD_mc  = (const float*)d_in[9];
    const float*     bias_mc  = (const float*)d_in[10];
    float* out = (float*)d_out;

    // workspace layout (floats)
    float* ws = (float*)d_ws;
    const long long NF  = (long long)N_NODES * OUT_HC;   // 6.4M
    const long long NH  = (long long)N_NODES * HEADS;    // 100K
    float* xw_pc  = ws;                 // NF
    float* xw_mc  = xw_pc  + NF;        // NF
    float* acc_pc = xw_mc  + NF;        // NF
    float* acc_mc = acc_pc + NF;        // NF
    float* as_pc  = acc_mc + NF;        // NH
    float* ad_pc  = as_pc  + NH;
    float* as_mc  = ad_pc  + NH;
    float* ad_mc  = as_mc  + NH;
    float* den_pc = ad_mc  + NH;
    float* den_mc = den_pc + NH;
    unsigned* m_pc = (unsigned*)(den_mc + NH);
    unsigned* m_mc = m_pc + NH;
    __bf16* wt = (__bf16*)(m_mc + NH);  // 2 * 128 * 128 bf16 = 64KB

    // K0a: init accumulators (every call; harness poisons ws only once)
    init_kernel<<<dim3(2048), dim3(256), 0, stream>>>(
        acc_pc, acc_mc, den_pc, den_mc, m_pc, m_mc);

    // K0b: W -> transposed bf16
    wt_kernel<<<dim3(OUT_HC, 2), dim3(IN_C), 0, stream>>>(W_pc, W_mc, wt);

    // K1: xw = x @ W (WMMA bf16), both convs via gridDim.y
    gemm_xw_kernel<<<dim3(N_NODES / 16, 2), dim3(256), 0, stream>>>(
        x, wt, xw_pc, xw_mc);

    // K2: attention logits per node/head
    att_kernel<<<dim3((N_NODES + 7) / 8, 2), dim3(256), 0, stream>>>(
        xw_pc, xw_mc, attS_pc, attD_pc, attS_mc, attD_mc,
        as_pc, ad_pc, as_mc, ad_mc);

    // K3: segment max
    edge_max_kernel<<<dim3((E_EDGES + 255) / 256, 2), dim3(256), 0, stream>>>(
        ei_pc, ei_mc, as_pc, ad_pc, as_mc, ad_mc, m_pc, m_mc);

    // K4: segment exp-sum
    edge_sum_kernel<<<dim3((E_EDGES + 255) / 256, 2), dim3(256), 0, stream>>>(
        ei_pc, ei_mc, as_pc, ad_pc, as_mc, ad_mc, m_pc, m_mc, den_pc, den_mc);

    // K5: weighted message scatter (wave per edge)
    edge_msg_kernel<<<dim3((E_EDGES + 7) / 8, 2), dim3(256), 0, stream>>>(
        ei_pc, ei_mc, as_pc, ad_pc, as_mc, ad_mc, m_pc, m_mc,
        den_pc, den_mc, xw_pc, xw_mc, acc_pc, acc_mc);

    // K6: bias + ELU + average
    final_kernel<<<dim3((int)(((long long)N_NODES * OUT_HC + 255) / 256)),
                   dim3(256), 0, stream>>>(acc_pc, acc_mc, bias_pc, bias_mc, out);
}
